// MultiHeadAttention_6090263626512
// MI455X (gfx1250) — matrix operand order repaired and checked by reading, not compiled, not run on hardware
//
#include <hip/hip_runtime.h>
#include <cstddef>
#include <cstdint>

#define DEVINL __device__ __forceinline__

typedef __attribute__((ext_vector_type(16))) __bf16 v16bf;
typedef __attribute__((ext_vector_type(8)))  float  v8f;
typedef __attribute__((ext_vector_type(4)))  unsigned int u32x4;
typedef __attribute__((ext_vector_type(2)))  unsigned int u32x2;
typedef __attribute__((ext_vector_type(8)))  int          i32x8;
typedef __attribute__((ext_vector_type(4)))  int          i32x4;
typedef unsigned short ushort_t;

union Frag16 { v16bf v; u32x4 q[2]; };

#if __has_builtin(__builtin_amdgcn_tensor_load_to_lds) && \
    __has_builtin(__builtin_amdgcn_s_wait_tensorcnt)
#define HAVE_TDM 1
#else
#define HAVE_TDM 0
#endif

// ---- helpers -------------------------------------------------------------

DEVINL ushort_t f2b(float f) {           // f32 -> bf16 (round to nearest even)
  union { float f; unsigned u; } cv; cv.f = f;
  unsigned u = cv.u;
  u += 0x7FFFu + ((u >> 16) & 1u);
  return (ushort_t)(u >> 16);
}
DEVINL unsigned pack2(float a, float b) {
  return (unsigned)f2b(a) | ((unsigned)f2b(b) << 16);
}

// A fragment (16x32 bf16): lanes 0-15 -> M=lane, K {0..7,16..23};
// lanes 16-31 -> K {8..15,24..31}. Two 16B LDS loads.
DEVINL v16bf load_frag_a(const ushort_t* tile, int lda, int m0, int kk, int lane) {
  int m  = m0 + (lane & 15);
  int kb = kk + ((lane & 16) ? 8 : 0);
  const ushort_t* p = tile + m * lda + kb;
  Frag16 f;
  f.q[0] = *(const u32x4*)(p);
  f.q[1] = *(const u32x4*)(p + 16);
  return f.v;
}
// B fragment (32x16 bf16): n = lane%16; lanes 0-15 -> K {0..7,16..23},
// lanes 16-31 -> K {8..15,24..31} (matches A).
DEVINL v16bf load_frag_b(const ushort_t* tile, int ldb, int kk, int n0, int lane) {
  const ushort_t* p = tile + (kk + ((lane & 16) ? 8 : 0)) * ldb + n0 + (lane & 15);
  union { v16bf v; ushort_t u[16]; } f;
  for (int s = 0; s < 8; ++s) {
    f.u[s]     = p[s * ldb];
    f.u[8 + s] = p[(16 + s) * ldb];
  }
  return f.v;
}
DEVINL v8f wmma_bf16(v16bf a, v16bf b, v8f c) {
  return __builtin_amdgcn_wmma_f32_16x16x32_bf16(
      false, a, false, b, (short)0, c, false, false);
}
DEVINL v8f zero8() { v8f z; for (int i = 0; i < 8; ++i) z[i] = 0.0f; return z; }

#if HAVE_TDM
// TDM: load a 2D bf16 tile [rows x width] (row stride `stride` elements) from
// global into LDS at byte offset lds_addr, padding each row to match the
// padded LDS stride. pad_int: row bytes = 8 * 2^pad_int; pad_amt: pad DWORDs-1.
DEVINL void tdm_load_2d_bf16(unsigned lds_addr, const ushort_t* gptr,
                             unsigned width, unsigned rows, unsigned stride,
                             unsigned pad_int, unsigned pad_amt) {
  unsigned long long ga = (unsigned long long)(uintptr_t)gptr;
  u32x4 g0;
  g0[0] = 1u;                                            // count=1, user mode
  g0[1] = lds_addr;                                      // LDS byte address
  g0[2] = (unsigned)(ga & 0xffffffffu);                  // global_addr[31:0]
  g0[3] = (unsigned)((ga >> 32) & 0x01ffffffu)           // global_addr[56:32]
        | 0x80000000u;                                   // type=2 ("image")
  i32x8 g1;
  g1[0] = (int)((1u << 16)            // data_size = 1 -> 2 bytes
              | (1u << 20)            // pad_enable
              | (pad_int << 22)       // pad_interval
              | (pad_amt << 25));     // pad_amount
  g1[1] = (int)(width << 16);                      // tensor_dim0[15:0] @63:48
  g1[2] = (int)((width >> 16) | (rows << 16));     // dim0 hi | tensor_dim1 lo
  g1[3] = (int)((rows  >> 16) | (width << 16));    // dim1 hi | tile_dim0
  g1[4] = (int)rows;                               // tile_dim1 (tile_dim2=0)
  g1[5] = (int)stride;                             // tensor_dim0_stride[31:0]
  g1[6] = 0;                                       // stride hi | dim1_stride lo
  g1[7] = 0;
  i32x4 z4; z4[0] = z4[1] = z4[2] = z4[3] = 0;
  i32x8 z8; for (int i = 0; i < 8; ++i) z8[i] = 0;
  // amdgpu-toolchain (clang-23) 6-arg form.
  __builtin_amdgcn_tensor_load_to_lds(g0, g1, z4, z4, z8, 0);
}
#endif

// ---- kernel 1/5: M=4096, N=1024, K=1024 GEMM (QKV projections + out proj) -
// OUT_MODE: 0 = bf16 [bh][n][64] (Q,V)   1 = bf16 [bh][64][2048] (K^T)
//           3 = f32 row-major [4096][1024] (final output)
template<bool A_IS_F32, int OUT_MODE>
__global__ __launch_bounds__(256) void gemm1024(
    const void* __restrict__ Aptr, const float* __restrict__ Bw,
    const float* __restrict__ bias, void* __restrict__ Out)
{
  constexpr int Ncol = 1024, K = 1024;
  constexpr int BM = 64, BN = 128, BK = 64, LDA = 72, LDB = 136;
  __shared__ __align__(16) ushort_t As[BM * LDA];
  __shared__ __align__(16) ushort_t Bs[BK * LDB];
  const int tid = threadIdx.x;
  const int lane = tid & 31, wave = tid >> 5;     // wave32
  const int bm = blockIdx.y * BM;
  const int bn = blockIdx.x * BN;
  const int waveM = (wave >> 2) * 32;             // 8 waves: 2 x 4
  const int waveN = (wave & 3) * 32;
  v8f acc[2][2];
  acc[0][0] = zero8(); acc[0][1] = zero8();
  acc[1][0] = zero8(); acc[1][1] = zero8();

  const int ar = tid >> 2, ac = (tid & 3) * 16;   // A tile: 64x64
  const int br = tid >> 5, bc = (tid & 31) * 4;   // B tile: 64x128

  for (int k0 = 0; k0 < K; k0 += BK) {
    // A tile
    if (A_IS_F32) {
      const float* Af = (const float*)Aptr;
      for (int j = 0; j < 4; ++j) {
        float4 x = *(const float4*)(Af + (size_t)(bm + ar) * K + k0 + ac + j*4);
        u32x2 t; t.x = pack2(x.x, x.y); t.y = pack2(x.z, x.w);
        *(u32x2*)(&As[ar * LDA + ac + j*4]) = t;
      }
    } else {
      const ushort_t* Ab = (const ushort_t*)Aptr;
      const ushort_t* g = Ab + (size_t)(bm + ar) * K + k0 + ac;
      *(u32x4*)(&As[ar * LDA + ac])     = *(const u32x4*)(g);
      *(u32x4*)(&As[ar * LDA + ac + 8]) = *(const u32x4*)(g + 8);
    }
    // B tile (f32 weights -> bf16)
    for (int rr = br; rr < BK; rr += 8) {
      float4 x = *(const float4*)(Bw + (size_t)(k0 + rr) * Ncol + bn + bc);
      u32x2 t; t.x = pack2(x.x, x.y); t.y = pack2(x.z, x.w);
      *(u32x2*)(&Bs[rr * LDB + bc]) = t;
    }
    if (k0 + BK < K)  // prefetch next weight K-slab
      __builtin_prefetch(Bw + (size_t)(k0 + BK + br) * Ncol + bn + bc, 0, 0);
    __syncthreads();
    for (int kk = 0; kk < BK; kk += 32) {
      v16bf a0 = load_frag_a(As, LDA, waveM + 0,  kk, lane);
      v16bf a1 = load_frag_a(As, LDA, waveM + 16, kk, lane);
      v16bf b0 = load_frag_b(Bs, LDB, kk, waveN + 0,  lane);
      v16bf b1 = load_frag_b(Bs, LDB, kk, waveN + 16, lane);
      acc[0][0] = wmma_bf16(a0, b0, acc[0][0]);
      acc[0][1] = wmma_bf16(a0, b1, acc[0][1]);
      acc[1][0] = wmma_bf16(a1, b0, acc[1][0]);
      acc[1][1] = wmma_bf16(a1, b1, acc[1][1]);
    }
    __syncthreads();
  }

  const int ln = lane & 15;
  const int mhi = (lane & 16) ? 8 : 0;
  for (int im = 0; im < 2; ++im)
    for (int in = 0; in < 2; ++in) {
      v8f c = acc[im][in];
      int gn = bn + waveN + in * 16 + ln;
      float bv = bias[gn];
      for (int r = 0; r < 8; ++r) {
        int gm = bm + waveM + im * 16 + mhi + r;
        float val = c[r] + bv;
        if (OUT_MODE == 3) {
          ((float*)Out)[(size_t)gm * Ncol + gn] = val;
        } else {
          int b = gm >> 11, n = gm & 2047;
          int h = gn >> 6,  d = gn & 63;
          ushort_t bb = f2b(val);
          if (OUT_MODE == 1)
            ((ushort_t*)Out)[(((size_t)(b * 16 + h)) * 64 + d) * 2048 + n] = bb;
          else
            ((ushort_t*)Out)[(((size_t)(b * 16 + h)) * 2048 + n) * 64 + d] = bb;
        }
      }
    }
}

// ---- kernel 2: scores S = 0.125 * Q K^T per (b,h), K=64 in one shot ------
__global__ __launch_bounds__(256) void scores_kernel(
    const ushort_t* __restrict__ Q, const ushort_t* __restrict__ KT,
    float* __restrict__ S)
{
  constexpr int N = 2048, DK = 64;
  constexpr int BM = 64, BN = 128, LDA = 72, LDB = 136;
  __shared__ __align__(16) ushort_t As[BM * LDA];
  __shared__ __align__(16) ushort_t Bs[DK * LDB];
  const int tid = threadIdx.x, lane = tid & 31, wave = tid >> 5;
  const int bh = blockIdx.z;
  const int bm = blockIdx.y * BM;
  const int bn = blockIdx.x * BN;
  const int waveM = (wave >> 2) * 32;
  const int waveN = (wave & 3) * 32;
  const ushort_t* Qb = Q  + (size_t)bh * N * DK;
  const ushort_t* Kb = KT + (size_t)bh * DK * N;
  v8f acc[2][2];
  acc[0][0] = zero8(); acc[0][1] = zero8();
  acc[1][0] = zero8(); acc[1][1] = zero8();

#if HAVE_TDM
  // Two concurrent DMAs: each issuing wave has its own TENSORcnt.
  if (wave == 0) {
    // A: 64 rows x 64 elems contiguous (stride 64); row 128B -> pad_int 4,
    // pad 8 elems (4 DW) -> code 3 => LDS stride 72.
    tdm_load_2d_bf16((unsigned)(uintptr_t)(void*)As, Qb + (size_t)bm * DK,
                     64, 64, 64, 4, 3);
    __builtin_amdgcn_s_wait_tensorcnt((short)0);
  } else if (wave == 1) {
    // B: 64 rows x 128 elems, row stride 2048; row 256B -> pad_int 5,
    // pad 8 elems -> 3 => LDS stride 136.
    tdm_load_2d_bf16((unsigned)(uintptr_t)(void*)Bs, Kb + bn,
                     128, 64, 2048, 5, 3);
    __builtin_amdgcn_s_wait_tensorcnt((short)0);
  }
  __syncthreads();
#else
  {
    const int ar = tid >> 2, ac = (tid & 3) * 16;
    const ushort_t* g = Qb + (size_t)(bm + ar) * DK + ac;
    *(u32x4*)(&As[ar * LDA + ac])     = *(const u32x4*)(g);
    *(u32x4*)(&As[ar * LDA + ac + 8]) = *(const u32x4*)(g + 8);
    const int brr = tid >> 4, bcc = (tid & 15) * 8;
    for (int rr = brr; rr < DK; rr += 16)
      *(u32x4*)(&Bs[rr * LDB + bcc]) =
          *(const u32x4*)(Kb + (size_t)rr * N + bn + bcc);
  }
  __syncthreads();
#endif

  for (int kk = 0; kk < DK; kk += 32) {
    v16bf a0 = load_frag_a(As, LDA, waveM + 0,  kk, lane);
    v16bf a1 = load_frag_a(As, LDA, waveM + 16, kk, lane);
    v16bf b0 = load_frag_b(Bs, LDB, kk, waveN + 0,  lane);
    v16bf b1 = load_frag_b(Bs, LDB, kk, waveN + 16, lane);
    acc[0][0] = wmma_bf16(a0, b0, acc[0][0]);
    acc[0][1] = wmma_bf16(a0, b1, acc[0][1]);
    acc[1][0] = wmma_bf16(a1, b0, acc[1][0]);
    acc[1][1] = wmma_bf16(a1, b1, acc[1][1]);
  }

  const int ln = lane & 15, mhi = (lane & 16) ? 8 : 0;
  for (int im = 0; im < 2; ++im)
    for (int in = 0; in < 2; ++in) {
      int gn = bn + waveN + in * 16 + ln;
      for (int r = 0; r < 8; ++r) {
        int gm = bm + waveM + im * 16 + mhi + r;
        S[((size_t)bh * N + gm) * N + gn] = acc[im][in][r] * 0.125f;
      }
    }
}

// ---- kernel 3: row softmax over 2048, in place ---------------------------
__global__ __launch_bounds__(256) void softmax_kernel(float* __restrict__ A)
{
  const int row = blockIdx.x;
  float* p = A + (size_t)row * 2048;
  const int tid = threadIdx.x;
  const int lane = tid & 31, wave = tid >> 5;
  __shared__ float red[8];

  float4 x0 = *(float4*)(p + tid * 8);
  float4 x1 = *(float4*)(p + tid * 8 + 4);
  float v[8] = { x0.x, x0.y, x0.z, x0.w, x1.x, x1.y, x1.z, x1.w };
  float m = v[0];
  for (int i = 1; i < 8; ++i) m = fmaxf(m, v[i]);
  for (int off = 16; off; off >>= 1) m = fmaxf(m, __shfl_xor(m, off, 32));
  if (lane == 0) red[wave] = m;
  __syncthreads();
  float rm = red[0];
  for (int i = 1; i < 8; ++i) rm = fmaxf(rm, red[i]);

  float s = 0.0f;
  for (int i = 0; i < 8; ++i) { v[i] = __expf(v[i] - rm); s += v[i]; }
  for (int off = 16; off; off >>= 1) s += __shfl_xor(s, off, 32);
  __syncthreads();
  if (lane == 0) red[wave] = s;
  __syncthreads();
  float rs = 0.0f;
  for (int i = 0; i < 8; ++i) rs += red[i];
  float inv = 1.0f / rs;

  *(float4*)(p + tid * 8)     = make_float4(v[0]*inv, v[1]*inv, v[2]*inv, v[3]*inv);
  *(float4*)(p + tid * 8 + 4) = make_float4(v[4]*inv, v[5]*inv, v[6]*inv, v[7]*inv);
}

// ---- kernel 4: Z = alpha @ V per (b,h), out bf16 [b][n][h*64+d] ----------
// Double-buffered TDM pipeline on the V tiles: wave 0 issues tile it+1's DMA,
// then s_wait_tensorcnt 1 (in-order per-wave counter) guarantees tile it is
// resident, overlapping the DMA with the alpha f32->bf16 staging + WMMAs.
__global__ __launch_bounds__(256) void zgemm_kernel(
    const float* __restrict__ S, const ushort_t* __restrict__ V,
    ushort_t* __restrict__ Z)
{
  constexpr int N = 2048, DV = 64;
  constexpr int BM = 128, BK = 64, LDA = 72, LDB = 72;
  constexpr int NITER = N / BK;                   // 32
  constexpr int BUFE = BK * LDB;                  // elements per V buffer
  __shared__ __align__(16) ushort_t As[BM * LDA];
  __shared__ __align__(16) ushort_t Bs[2 * BUFE];
  const int tid = threadIdx.x, lane = tid & 31, wave = tid >> 5;
  const int bh = blockIdx.z;
  const int bm = blockIdx.y * BM;
  const int waveM = (wave >> 1) * 32;             // 8 waves: 4 x 2
  const int waveN = (wave & 1) * 32;
  const float*    Sb = S + (size_t)bh * N * N;
  const ushort_t* Vb = V + (size_t)bh * N * DV;
  v8f acc[2][2];
  acc[0][0] = zero8(); acc[0][1] = zero8();
  acc[1][0] = zero8(); acc[1][1] = zero8();
  const int ar = tid >> 2, ac = (tid & 3) * 16;   // A tile 128x64 (2 passes)

#if HAVE_TDM
  const unsigned ldsB = (unsigned)(uintptr_t)(void*)Bs;
  if (wave == 0)                                   // prologue: tile 0 -> buf 0
    tdm_load_2d_bf16(ldsB, Vb, 64, 64, 64, 4, 3);
#endif

  for (int it = 0; it < NITER; ++it) {
    const int k0 = it * BK;
    ushort_t* Bcur = &Bs[(it & 1) * BUFE];
#if HAVE_TDM
    if (wave == 0 && it + 1 < NITER)               // issue next V tile
      tdm_load_2d_bf16(ldsB + (unsigned)(((it + 1) & 1) * BUFE * 2),
                       Vb + (size_t)(k0 + BK) * DV, 64, 64, 64, 4, 3);
#else
    {
      const ushort_t* g = Vb + (size_t)(k0 + ar) * DV + ac;
      *(u32x4*)(&Bcur[ar * LDB + ac])     = *(const u32x4*)(g);
      *(u32x4*)(&Bcur[ar * LDB + ac + 8]) = *(const u32x4*)(g + 8);
    }
#endif
    // alpha tile f32 -> bf16 (overlaps with in-flight DMA)
    for (int rr = ar; rr < BM; rr += 64) {
      for (int j = 0; j < 4; ++j) {
        float4 x = *(const float4*)(Sb + (size_t)(bm + rr) * N + k0 + ac + j*4);
        u32x2 t; t.x = pack2(x.x, x.y); t.y = pack2(x.z, x.w);
        *(u32x2*)(&As[rr * LDA + ac + j*4]) = t;
      }
    }
#if HAVE_TDM
    if (wave == 0) {
      if (it + 1 < NITER) __builtin_amdgcn_s_wait_tensorcnt((short)1);
      else                __builtin_amdgcn_s_wait_tensorcnt((short)0);
    }
#endif
    __syncthreads();
    for (int kk = 0; kk < BK; kk += 32) {
      v16bf a0 = load_frag_a(As, LDA, waveM + 0,  kk, lane);
      v16bf a1 = load_frag_a(As, LDA, waveM + 16, kk, lane);
      v16bf b0 = load_frag_b(Bcur, LDB, kk, waveN + 0,  lane);
      v16bf b1 = load_frag_b(Bcur, LDB, kk, waveN + 16, lane);
      acc[0][0] = wmma_bf16(a0, b0, acc[0][0]);
      acc[0][1] = wmma_bf16(a0, b1, acc[0][1]);
      acc[1][0] = wmma_bf16(a1, b0, acc[1][0]);
      acc[1][1] = wmma_bf16(a1, b1, acc[1][1]);
    }
    __syncthreads();
  }

  const int ln = lane & 15, mhi = (lane & 16) ? 8 : 0;
  const int b = bh >> 4, h = bh & 15;
  for (int im = 0; im < 2; ++im)
    for (int in = 0; in < 2; ++in) {
      int d = waveN + in * 16 + ln;
      for (int r = 0; r < 8; ++r) {
        int mq = bm + waveM + im * 16 + mhi + r;
        Z[((size_t)(b * 2048 + mq)) * 1024 + h * 64 + d] = f2b(acc[im][in][r]);
      }
    }
}

// ---- host launcher -------------------------------------------------------

extern "C" void kernel_launch(void* const* d_in, const int* in_sizes, int n_in,
                              void* d_out, int out_size, void* d_ws, size_t ws_size,
                              hipStream_t stream) {
  (void)in_sizes; (void)n_in; (void)out_size; (void)ws_size;
  const float* X  = (const float*)d_in[0];
  const float* Wq = (const float*)d_in[1];
  const float* bq = (const float*)d_in[2];
  const float* Wk = (const float*)d_in[3];
  const float* bk = (const float*)d_in[4];
  const float* Wv = (const float*)d_in[5];
  const float* bv = (const float*)d_in[6];
  const float* Wo = (const float*)d_in[7];
  const float* bo = (const float*)d_in[8];

  float* out   = (float*)d_out;
  float* alpha = out + (size_t)4096 * 1024;              // [32][2048][2048] f32

  const size_t QKV_ELEMS = (size_t)32 * 2048 * 64;       // bf16 elements each
  ushort_t* Qbf  = (ushort_t*)d_ws;
  ushort_t* KbfT = Qbf  + QKV_ELEMS;
  ushort_t* Vbf  = KbfT + QKV_ELEMS;
  ushort_t* Zbf  = Vbf  + QKV_ELEMS;                     // total 32 MB of ws

  dim3 blk(256);
  dim3 gProj(8, 64);        // N=1024/128, M=4096/64
  dim3 gScore(16, 32, 32);  // N=2048/128, M=2048/64, bh=32
  dim3 gZ(1, 16, 32);       // N=64/64,   M=2048/128, bh=32

  gemm1024<true, 0><<<gProj, blk, 0, stream>>>(X, Wq, bq, (void*)Qbf);
  gemm1024<true, 1><<<gProj, blk, 0, stream>>>(X, Wk, bk, (void*)KbfT);
  gemm1024<true, 0><<<gProj, blk, 0, stream>>>(X, Wv, bv, (void*)Vbf);
  scores_kernel<<<gScore, blk, 0, stream>>>(Qbf, KbfT, alpha);
  softmax_kernel<<<dim3(32 * 2048), blk, 0, stream>>>(alpha);
  zgemm_kernel<<<gZ, blk, 0, stream>>>(alpha, Vbf, Zbf);
  gemm1024<false, 3><<<gProj, blk, 0, stream>>>(Zbf, Wo, bo, (void*)out);
}
